// MultiHeadAttention_37855841747247
// MI455X (gfx1250) — matrix operand order repaired and checked by reading, not compiled, not run on hardware
//
#include <hip/hip_runtime.h>
#include <hip/hip_bf16.h>

// MI455X (gfx1250) multi-head attention: f32 inputs -> f16 WMMA path, f32 accum.
// Stage 1: projection GEMMs (X@W for Q,K,V) via v_wmma_f32_16x16x32_f16,
//          double-buffered LDS tiles, LDS-staged vectorized epilogue.
// Stage 2: flash attention; scores transposed once through LDS, softmax done
//          row-per-lane (2 shuffles/tile instead of 64), P fragment built
//          directly in registers.

typedef _Float16 f16;
typedef __attribute__((ext_vector_type(16))) _Float16 v16h;
typedef __attribute__((ext_vector_type(8)))  _Float16 v8h;
typedef __attribute__((ext_vector_type(4)))  _Float16 v4h;
typedef __attribute__((ext_vector_type(8)))  float    v8f;

#define NB   8      // batch
#define NS   1024   // sequence
#define ND   1024   // model dim
#define NH   16     // heads
#define HD   64     // head dim
#define LDA  40     // padded LDS stride (halfwords) for 128x32 tiles

union V16 { v16h v; v8h h[2]; };

__device__ __forceinline__ v8f v8f_zero() {
  v8f z = {0.f, 0.f, 0.f, 0.f, 0.f, 0.f, 0.f, 0.f};
  return z;
}

__device__ __forceinline__ v8f wmma_f16(v16h a, v16h b, v8f c) {
  // (neg_a, A, neg_b, B, c_mod, C, reuse_a, reuse_b)
  return __builtin_amdgcn_wmma_f32_16x16x32_f16(false, a, false, b, (short)0, c,
                                                false, false);
}

// ---------------------------------------------------------------------------
// Stage 1: projection GEMM. M = B*S = 8192, N = 1024, K = 1024.
// Block tile 128x128, 8 waves (4x2), wave tile 32x64 = 2x4 WMMA accumulators.
// WHICH 0/1: out [b][h][s][64] (Q/K)    WHICH 2: out [b][h][d][s] (V^T)
// ---------------------------------------------------------------------------
template <int WHICH>
__global__ __launch_bounds__(256)
void proj_kernel(const float* __restrict__ x, const float* __restrict__ w,
                 f16* __restrict__ o)
{
  __shared__ f16 lds_a[2][128 * LDA];   // A tile  [m][k], 128x32, double buffered
  __shared__ f16 lds_bt[2][128 * LDA];  // B tile transposed [n][k], 128x32
  // Epilogue staging: Q/K -> [m_local][n_local] 32x64 (stride 72);
  //                   V^T -> [n_local][m_local] 64x32 (stride 40).
  __shared__ f16 lds_c[8][(WHICH == 2) ? (64 * 40) : (32 * 72)];

  const int tid  = threadIdx.x;
  const int lane = tid & 31;
  const int wid  = tid >> 5;
  const int wm   = wid >> 1;          // 0..3  (M direction)
  const int wn   = wid & 1;           // 0..1  (N direction)
  const int m0   = blockIdx.y * 128;
  const int n0   = blockIdx.x * 128;
  const int c16  = lane & 15;
  const int kb   = (lane >> 4) * 8;   // A fragment K base
  const int kb2  = (lane >> 4) * 16;  // unused here; B fragment uses kb too
  const int rowoff = (lane >> 4) * 8;

  v8f acc[2][4];
  #pragma unroll
  for (int i = 0; i < 2; ++i)
    #pragma unroll
    for (int j = 0; j < 4; ++j) acc[i][j] = v8f_zero();

  // Stage one 128x32 A tile and 32x128 B tile (as [n][k]) into buffer `buf`.
  auto stage = [&](int buf, int k0) {
    #pragma unroll
    for (int i = 0; i < 4; ++i) {
      int slot = tid + i * 256;          // 1024 float4 slots
      int row  = slot >> 3;
      int c4   = slot & 7;
      const float4 vv =
          *(const float4*)(x + (size_t)(m0 + row) * ND + k0 + c4 * 4);
      v4h hv = {(f16)vv.x, (f16)vv.y, (f16)vv.z, (f16)vv.w};
      *(v4h*)(&lds_a[buf][row * LDA + c4 * 4]) = hv;
    }
    #pragma unroll
    for (int i = 0; i < 4; ++i) {
      int slot = tid + i * 256;
      int kk   = slot >> 5;
      int c4   = slot & 31;
      const float4 vv =
          *(const float4*)(w + (size_t)(k0 + kk) * ND + n0 + c4 * 4);
      int n = c4 * 4;
      lds_bt[buf][(n + 0) * LDA + kk] = (f16)vv.x;
      lds_bt[buf][(n + 1) * LDA + kk] = (f16)vv.y;
      lds_bt[buf][(n + 2) * LDA + kk] = (f16)vv.z;
      lds_bt[buf][(n + 3) * LDA + kk] = (f16)vv.w;
    }
  };

  auto compute = [&](int buf) {
    V16 af[2];
    #pragma unroll
    for (int sm = 0; sm < 2; ++sm) {
      int row = wm * 32 + sm * 16 + c16;
      af[sm].h[0] = *(const v8h*)(&lds_a[buf][row * LDA + kb]);
      af[sm].h[1] = *(const v8h*)(&lds_a[buf][row * LDA + kb + 16]);
    }
    #pragma unroll
    for (int sn = 0; sn < 4; ++sn) {
      int col = wn * 64 + sn * 16 + c16;
      V16 bf;
      bf.h[0] = *(const v8h*)(&lds_bt[buf][col * LDA + kb]);
      bf.h[1] = *(const v8h*)(&lds_bt[buf][col * LDA + kb + 16]);
      #pragma unroll
      for (int sm = 0; sm < 2; ++sm)
        acc[sm][sn] = wmma_f16(af[sm].v, bf.v, acc[sm][sn]);
    }
  };

  // Double-buffered K loop: one barrier per iteration.
  stage(0, 0);
  __syncthreads();
  for (int k0 = 0; k0 < ND; k0 += 32) {
    const int cur = (k0 >> 5) & 1;
    if (k0 + 32 < ND) stage(cur ^ 1, k0 + 32);
    compute(cur);
    __syncthreads();
  }

  // ---------------- Epilogue via LDS restage -> coalesced b128 stores -------
  f16* cs = &lds_c[wid][0];
  if constexpr (WHICH != 2) {
    #pragma unroll
    for (int sm = 0; sm < 2; ++sm)
      #pragma unroll
      for (int sn = 0; sn < 4; ++sn)
        #pragma unroll
        for (int r = 0; r < 8; ++r)
          cs[(sm * 16 + rowoff + r) * 72 + sn * 16 + c16] =
              (f16)acc[sm][sn][r];
    __asm volatile("s_wait_dscnt 0" ::: "memory");  // wave-local transpose

    const int mrow = m0 + wm * 32 + lane;   // each lane emits one output row
    const int b = mrow >> 10, s = mrow & 1023;
    const int h = (n0 + wn * 64) >> 6;      // wave column span == one head
    f16* __restrict__ dst = o + (((size_t)b * NH + h) * NS + s) * HD;
    #pragma unroll
    for (int j = 0; j < 8; ++j)
      *(v8h*)(dst + j * 8) = *(const v8h*)(&cs[lane * 72 + j * 8]);
  } else {
    #pragma unroll
    for (int sm = 0; sm < 2; ++sm)
      #pragma unroll
      for (int sn = 0; sn < 4; ++sn)
        #pragma unroll
        for (int r = 0; r < 8; ++r)
          cs[(sn * 16 + c16) * 40 + sm * 16 + rowoff + r] =
              (f16)acc[sm][sn][r];
    __asm volatile("s_wait_dscnt 0" ::: "memory");

    const int b  = m0 >> 10;
    const int s0 = (m0 & 1023) + wm * 32;   // 32 contiguous seq positions
    const int h  = (n0 + wn * 64) >> 6;
    #pragma unroll
    for (int t = 0; t < 2; ++t) {
      const int dl = t * 32 + lane;         // head-dim row 0..63
      f16* __restrict__ dst = o + (((size_t)b * NH + h) * HD + dl) * NS + s0;
      #pragma unroll
      for (int j = 0; j < 4; ++j)
        *(v8h*)(dst + j * 8) = *(const v8h*)(&cs[dl * 40 + j * 8]);
    }
  }
}

// ---------------------------------------------------------------------------
// Stage 2: flash attention. One block per (128 q-rows, head, batch);
// each wave owns 16 q-rows. Key loop in tiles of 32.
// Softmax stats are kept row-per-lane: lane L and L+16 jointly own score
// row (L&15), so reductions need a single xor-16 shuffle.
// ---------------------------------------------------------------------------
__global__ __launch_bounds__(256)
void attn_kernel(const f16* __restrict__ q, const f16* __restrict__ k,
                 const f16* __restrict__ vt, float* __restrict__ out)
{
  // Per-wave 16x32 f32 score scratch (stride 68), reused for output staging.
  __shared__ float lds_s[8][16 * 68];

  const int tid  = threadIdx.x;
  const int lane = tid & 31;
  const int wid  = tid >> 5;
  const int h    = blockIdx.y;
  const int b    = blockIdx.z;
  const int bh   = b * NH + h;
  const int qbase = blockIdx.x * 128 + wid * 16;

  const int c16    = lane & 15;
  const int kb     = (lane >> 4) * 8;
  const int kb2    = (lane >> 4) * 16;
  const int rowoff = (lane >> 4) * 8;
  const float scale = 0.125f;  // 1/sqrt(64)

  // Preload Q fragments: 16x64 per wave -> two 16x32 A-fragments.
  V16 qf[2];
  {
    const f16* qrow = q + ((size_t)bh * NS + qbase + c16) * HD;
    #pragma unroll
    for (int dstep = 0; dstep < 2; ++dstep) {
      qf[dstep].h[0] = *(const v8h*)(qrow + dstep * 32 + kb2);
      qf[dstep].h[1] = *(const v8h*)(qrow + dstep * 32 + kb2 + 8);
    }
  }

  v8f oacc[4];
  #pragma unroll
  for (int n = 0; n < 4; ++n) oacc[n] = v8f_zero();
  float m_cur = -3.0e38f;   // running max of the lane's owned row
  float l_cur = 0.f;        // running sum of the lane's owned row

  float* sb = &lds_s[wid][0];

  for (int key0 = 0; key0 < NS; key0 += 32) {
    // S tile 16x32 = two 16x16 accumulators; contract over head_dim (2 steps).
    v8f sacc[2];
    sacc[0] = v8f_zero();
    sacc[1] = v8f_zero();
    #pragma unroll
    for (int dstep = 0; dstep < 2; ++dstep) {
      #pragma unroll
      for (int nt = 0; nt < 2; ++nt) {
        const v16h bf = *(const v16h*)(
            k + ((size_t)bh * NS + key0 + nt * 16 + c16) * HD + dstep * 32 + kb2);
        sacc[nt] = wmma_f16(qf[dstep].v, bf, sacc[nt]);
      }
    }

    // Transpose raw scores through LDS (C layout -> row-major).
    #pragma unroll
    for (int nt = 0; nt < 2; ++nt)
      #pragma unroll
      for (int r = 0; r < 8; ++r)
        sb[(rowoff + r) * 68 + nt * 16 + c16] = sacc[nt][r];
    __asm volatile("s_wait_dscnt 0" ::: "memory");  // wave-local transpose

    // Lane reads its A-fragment footprint of row c16: cols {kb2..kb2+15}.
    // Lanes L and L+16 cover the full 32 columns.
    float vals[16];
    {
      const float4 a0 = *(const float4*)(&sb[c16 * 68 + kb2]);
      const float4 a1 = *(const float4*)(&sb[c16 * 68 + kb2 + 4]);
      const float4 a2 = *(const float4*)(&sb[c16 * 68 + kb2 + 8]);
      const float4 a3 = *(const float4*)(&sb[c16 * 68 + kb2 + 12]);
      vals[0] = a0.x; vals[1] = a0.y; vals[2]  = a0.z; vals[3]  = a0.w;
      vals[4] = a1.x; vals[5] = a1.y; vals[6]  = a1.z; vals[7]  = a1.w;
      vals[8] = a2.x; vals[9] = a2.y; vals[10] = a2.z; vals[11] = a2.w;
      vals[12] = a3.x; vals[13] = a3.y; vals[14] = a3.z; vals[15] = a3.w;
    }
    #pragma unroll
    for (int j = 0; j < 16; ++j) vals[j] *= scale;

    // Row max: local 16 + one xor-16 shuffle.
    float lmax = vals[0];
    #pragma unroll
    for (int j = 1; j < 16; ++j) lmax = fmaxf(lmax, vals[j]);
    lmax = fmaxf(lmax, __shfl_xor(lmax, 16, 32));

    const float mnew = fmaxf(m_cur, lmax);
    const float corr = __expf(m_cur - mnew);
    m_cur = mnew;

    // Exponentiate; P fragment assembled directly in A-fragment order.
    V16 pf;
    float lsum = 0.f;
    #pragma unroll
    for (int j = 0; j < 8; ++j) {
      float p0 = __expf(vals[j] - mnew);
      float p1 = __expf(vals[8 + j] - mnew);
      lsum += p0 + p1;
      pf.h[0][j] = (f16)p0;
      pf.h[1][j] = (f16)p1;
    }
    lsum += __shfl_xor(lsum, 16, 32);
    l_cur = l_cur * corr + lsum;

    // Broadcast per-row correction back to the C-layout accumulators.
    #pragma unroll
    for (int r = 0; r < 8; ++r) {
      const float cr = __shfl(corr, rowoff + r, 32);
      #pragma unroll
      for (int n = 0; n < 4; ++n) oacc[n][r] *= cr;
    }

    // O += P * V  (V^T rows give contiguous keys).
    #pragma unroll
    for (int n = 0; n < 4; ++n) {
      const v16h vf = *(const v16h*)(
          vt + ((size_t)bh * HD + n * 16 + c16) * NS + key0 + kb2);
      oacc[n] = wmma_f16(pf.v, vf, oacc[n]);
    }
  }

  // Normalize, restage through LDS, write out [b][s][h*64+d] as float4s.
  const float linv = 1.0f / l_cur;
  float lr[8];
  #pragma unroll
  for (int r = 0; r < 8; ++r) lr[r] = __shfl(linv, rowoff + r, 32);
  #pragma unroll
  for (int n = 0; n < 4; ++n)
    #pragma unroll
    for (int r = 0; r < 8; ++r)
      sb[(rowoff + r) * 68 + n * 16 + c16] = oacc[n][r] * lr[r];
  __asm volatile("s_wait_dscnt 0" ::: "memory");

  const int row  = lane >> 1;        // 16 rows, 2 lanes per row
  const int half = lane & 1;         // each lane covers 32 floats (128B)
  const float* srcrow = &sb[row * 68 + half * 32];
  float* __restrict__ dst =
      out + ((size_t)b * NS + qbase + row) * ND + h * HD + half * 32;
  #pragma unroll
  for (int j = 0; j < 8; ++j)
    *(float4*)(dst + j * 4) = *(const float4*)(srcrow + j * 4);
}

// ---------------------------------------------------------------------------
extern "C" void kernel_launch(void* const* d_in, const int* in_sizes, int n_in,
                              void* d_out, int out_size, void* d_ws, size_t ws_size,
                              hipStream_t stream) {
  const float* xq = (const float*)d_in[0];
  const float* xk = (const float*)d_in[1];
  const float* xv = (const float*)d_in[2];
  const float* wq = (const float*)d_in[3];
  const float* wk = (const float*)d_in[4];
  const float* wv = (const float*)d_in[5];
  float* out = (float*)d_out;

  // Workspace: Q, K as f16 [b][h][s][64]; V transposed f16 [b][h][d][s].
  // 3 * 8*16*1024*64 * 2B = ~50.3 MB.
  const size_t per = (size_t)NB * NH * NS * HD;
  f16* qbuf  = (f16*)d_ws;
  f16* kbuf  = qbuf + per;
  f16* vtbuf = kbuf + per;

  dim3 pgrid(ND / 128, (NB * NS) / 128, 1);
  proj_kernel<0><<<pgrid, 256, 0, stream>>>(xq, wq, qbuf);
  proj_kernel<1><<<pgrid, 256, 0, stream>>>(xk, wk, kbuf);
  proj_kernel<2><<<pgrid, 256, 0, stream>>>(xv, wv, vtbuf);

  dim3 agrid(NS / 128, NH, NB);
  attn_kernel<<<agrid, 256, 0, stream>>>(qbuf, kbuf, vtbuf, out);
}
